// MyQuantizedSelfAttentionLayer_25744033972528
// MI455X (gfx1250) — matrix operand order repaired and checked by reading, not compiled, not run on hardware
//
#include <hip/hip_runtime.h>
#include <cstdint>

#define D 256
#define SEQ 2048
#define BATCH 4

typedef __attribute__((ext_vector_type(16))) __bf16 v16bf;
typedef __attribute__((ext_vector_type(8)))  float  v8f;

union Op {
    uint32_t u[8];
    uint4    q[2];
    v16bf    v;
};

// --- f32 -> bf16 via hardware cvt (RNE) ---
__device__ __forceinline__ uint32_t pack_bf2(float lo, float hi) {
#if __has_builtin(__builtin_amdgcn_cvt_pk_bf16_f32)
    typedef __attribute__((ext_vector_type(2))) __bf16 v2bf;
    v2bf r = __builtin_amdgcn_cvt_pk_bf16_f32(lo, hi);
    return __builtin_bit_cast(uint32_t, r);
#else
    union { __bf16 h[2]; uint32_t u; } t;
    t.h[0] = (__bf16)lo;
    t.h[1] = (__bf16)hi;
    return t.u;
#endif
}
__device__ __forceinline__ uint16_t f2bf(float f) {
    return __builtin_bit_cast(uint16_t, (__bf16)f);
}

// A-operand (16x32 MxK, bf16): lane<16 -> K {kb..kb+7, kb+16..kb+23}; lane>=16 complementary.
__device__ __forceinline__ Op load_a_bf16(const uint16_t* m, int ld, int row0, int kb, int lane) {
    int row = row0 + (lane & 15);
    int sel = (lane >> 4) & 1;
    const uint16_t* r = m + (size_t)row * ld + kb;
    Op o;
    o.q[0] = *reinterpret_cast<const uint4*>(r + 8 * sel);
    o.q[1] = *reinterpret_cast<const uint4*>(r + 16 + 8 * sel);
    return o;
}
// B-operand (32x16 KxN, bf16): lane<16 -> K {kb..kb+7, kb+16..kb+23}; lane>=16 complementary, column = lane&15.
__device__ __forceinline__ Op load_b_bf16(const uint16_t* m, int ld, int col0, int kb, int lane) {
    int col = col0 + (lane & 15);
    int sel = (lane >> 4) & 1;
    const uint16_t* r = m + (size_t)col * ld + kb;
    Op o;
    o.q[0] = *reinterpret_cast<const uint4*>(r + 8 * sel);
    o.q[1] = *reinterpret_cast<const uint4*>(r + 16 + 8 * sel);
    return o;
}
__device__ __forceinline__ v8f wmma_bf16(const Op& a, const Op& b, v8f c) {
    return __builtin_amdgcn_wmma_f32_16x16x32_bf16(false, a.v, false, b.v, (short)0, c, false, false);
}

// ---------------- Kernel 0: bulk f32 -> bf16 convert (8 elements / thread) ----------------
__global__ void __launch_bounds__(256) cvt_kernel(const float* __restrict__ in,
                                                  uint16_t* __restrict__ out, int n8)
{
    int i = blockIdx.x * 256 + threadIdx.x;
    if (i >= n8) return;
    const float4* p = reinterpret_cast<const float4*>(in) + 2 * (size_t)i;
    float4 a = p[0], b = p[1];
    uint4 r;
    r.x = pack_bf2(a.x, a.y);
    r.y = pack_bf2(a.z, a.w);
    r.z = pack_bf2(b.x, b.y);
    r.w = pack_bf2(b.z, b.w);
    reinterpret_cast<uint4*>(out)[i] = r;
}

// ---------------- Kernel 1: QKV projection (x @ W^T + b), all-bf16 GEMM ----------------
// grid = (512 row-tiles, 4, 3 mats), block = 128 (4 waves). Wave -> one 16x16 tile.
// mat 0 -> K, 1 -> Q, 2 -> V (V stored transposed: Vt[b][d][s]).
__global__ void __launch_bounds__(128) qkv_proj_kernel(
    const uint16_t* __restrict__ xb, const uint16_t* __restrict__ Wb,
    const float* __restrict__ bk, const float* __restrict__ bq, const float* __restrict__ bv,
    uint16_t* __restrict__ Kb, uint16_t* __restrict__ Qb, uint16_t* __restrict__ Vt)
{
    int lane = threadIdx.x & 31;
    int wave = threadIdx.x >> 5;
    int rt   = blockIdx.x;                  // row tile over flat [B*S] rows
    int ct   = blockIdx.y * 4 + wave;       // col tile 0..15
    int mat  = blockIdx.z;
    const uint16_t* W  = Wb + (size_t)mat * D * D;
    const float* bias  = (mat == 0) ? bk : (mat == 1) ? bq : bv;
    int row0 = rt * 16;
    int n0   = ct * 16;

    v8f c = {};
#pragma unroll
    for (int t = 0; t < 8; ++t) {
        Op a = load_a_bf16(xb, D, row0, t * 32, lane);
        Op b = load_b_bf16(W,  D, n0,   t * 32, lane);
        c = wmma_bf16(a, b, c);
    }
    int n  = n0 + (lane & 15);
    int m0 = row0 + ((lane >> 4) & 1) * 8;
    float bn = bias[n];
    if (mat == 2) {
#pragma unroll
        for (int i = 0; i < 8; ++i) {
            int m = m0 + i;
            int bb = m >> 11;               // / SEQ
            int sl = m & (SEQ - 1);
            Vt[((size_t)bb * D + n) * SEQ + sl] = f2bf(c[i] + bn);
        }
    } else {
        uint16_t* out = (mat == 0) ? Kb : Qb;
#pragma unroll
        for (int i = 0; i < 8; ++i)
            out[(size_t)(m0 + i) * D + n] = f2bf(c[i] + bn);
    }
}

// ---------------- Kernel 2: scores = (Q K^T) / 16, raw f32 into attn region ----------------
// grid = (128 q-tiles, 4, 4 batches), block = 128. Each wave: one q-tile x 8 key-tiles,
// Q operands cached in registers across the key loop.
__global__ void __launch_bounds__(128) scores_kernel(
    const uint16_t* __restrict__ Qb, const uint16_t* __restrict__ Kb,
    float* __restrict__ attn)
{
    int lane = threadIdx.x & 31;
    int wave = threadIdx.x >> 5;
    int qt = blockIdx.x;
    int b  = blockIdx.z;
    int ktBase = (blockIdx.y * 4 + wave) * 8;   // 16 wave-slots * 8 = 128 key tiles
    const uint16_t* Q = Qb + (size_t)b * SEQ * D;
    const uint16_t* K = Kb + (size_t)b * SEQ * D;
    float* S = attn + (size_t)b * SEQ * SEQ;

    Op aops[8];
#pragma unroll
    for (int t = 0; t < 8; ++t)
        aops[t] = load_a_bf16(Q, D, qt * 16, t * 32, lane);

    int n_lo = lane & 15;
    int m0   = qt * 16 + ((lane >> 4) & 1) * 8;
    for (int kt = ktBase; kt < ktBase + 8; ++kt) {
        v8f c = {};
#pragma unroll
        for (int t = 0; t < 8; ++t) {
            Op bop = load_b_bf16(K, D, kt * 16, t * 32, lane);
            c = wmma_bf16(aops[t], bop, c);
        }
        int n = kt * 16 + n_lo;
#pragma unroll
        for (int i = 0; i < 8; ++i)
            S[(size_t)(m0 + i) * SEQ + n] = c[i] * 0.0625f;   // 1/sqrt(256)
    }
}

// ---------------- Kernel 3: row softmax in place; also emits bf16 copy of P ----------------
__global__ void __launch_bounds__(256) softmax_kernel(float* __restrict__ attn,
                                                      uint16_t* __restrict__ Pb)
{
    int row = blockIdx.x;                       // 0 .. B*SEQ-1
    float* p = attn + (size_t)row * SEQ;
    uint16_t* pb = Pb + (size_t)row * SEQ;
    int t = threadIdx.x;

    float v[8];
    float m = -3.402823466e38f;
#pragma unroll
    for (int i = 0; i < 8; ++i) {
        v[i] = p[t + i * 256];
        m = fmaxf(m, v[i]);
    }
#pragma unroll
    for (int off = 16; off > 0; off >>= 1)
        m = fmaxf(m, __shfl_xor(m, off, 32));

    __shared__ float red_max[8];
    __shared__ float red_sum[8];
    int wv = t >> 5;
    if ((t & 31) == 0) red_max[wv] = m;
    __syncthreads();
    m = red_max[0];
#pragma unroll
    for (int i = 1; i < 8; ++i) m = fmaxf(m, red_max[i]);

    float s = 0.f;
#pragma unroll
    for (int i = 0; i < 8; ++i) {
        v[i] = __expf(v[i] - m);
        s += v[i];
    }
#pragma unroll
    for (int off = 16; off > 0; off >>= 1)
        s += __shfl_xor(s, off, 32);
    if ((t & 31) == 0) red_sum[wv] = s;
    __syncthreads();
    s = red_sum[0];
#pragma unroll
    for (int i = 1; i < 8; ++i) s += red_sum[i];

    float inv = 1.0f / s;
#pragma unroll
    for (int i = 0; i < 8; ++i) {
        float w = v[i] * inv;
        p[t + i * 256]  = w;
        pb[t + i * 256] = f2bf(w);
    }
}

// ---------------- Kernel 4: output = P @ V, all-bf16 GEMM ----------------
// grid = (128 q-tiles, 4, 4 batches), block = 128. Wave -> one 16x16 (q,d) tile, 64 K-steps.
__global__ void __launch_bounds__(128) av_kernel(
    const uint16_t* __restrict__ Pb, const uint16_t* __restrict__ Vt,
    float* __restrict__ out)
{
    int lane = threadIdx.x & 31;
    int wave = threadIdx.x >> 5;
    int qt = blockIdx.x;
    int b  = blockIdx.z;
    int dt = blockIdx.y * 4 + wave;             // 0..15
    const uint16_t* P = Pb + (size_t)b * SEQ * SEQ;
    const uint16_t* V = Vt + (size_t)b * D * SEQ;

    v8f c = {};
    for (int t = 0; t < SEQ / 32; ++t) {
        Op a  = load_a_bf16(P, SEQ, qt * 16, t * 32, lane);
        Op bo = load_b_bf16(V, SEQ, dt * 16, t * 32, lane);
        c = wmma_bf16(a, bo, c);
    }
    int n  = dt * 16 + (lane & 15);
    int m0 = qt * 16 + ((lane >> 4) & 1) * 8;
#pragma unroll
    for (int i = 0; i < 8; ++i)
        out[((size_t)b * SEQ + m0 + i) * D + n] = c[i];
}

extern "C" void kernel_launch(void* const* d_in, const int* in_sizes, int n_in,
                              void* d_out, int out_size, void* d_ws, size_t ws_size,
                              hipStream_t stream) {
    const float* x  = (const float*)d_in[0];
    const float* Wk = (const float*)d_in[1];
    const float* bk = (const float*)d_in[2];
    const float* Wq = (const float*)d_in[3];
    const float* bq = (const float*)d_in[4];
    const float* Wv = (const float*)d_in[5];
    const float* bv = (const float*)d_in[6];

    float* out  = (float*)d_out;                       // [B][S][D]
    float* attn = out + (size_t)BATCH * SEQ * D;       // [B][S][S]

    const size_t NXD = (size_t)BATCH * SEQ * D;        // 2,097,152
    uint16_t* Kb = (uint16_t*)d_ws;                    // [B][S][D] bf16
    uint16_t* Qb = Kb + NXD;                           // [B][S][D] bf16
    uint16_t* Vt = Qb + NXD;                           // [B][D][S] bf16 (V transposed)
    uint16_t* xb = Vt + NXD;                           // [B][S][D] bf16 (x converted)
    uint16_t* Wb = xb + NXD;                           // 3 x [D][D] bf16 (Wk, Wq, Wv)
    uint16_t* Pb = Wb + (size_t)3 * D * D;             // [B][S][S] bf16 (softmax(P))

    // Pre-convert x and weights to bf16 (hoists all cvt out of the GEMMs).
    int n8x = (int)(NXD / 8);
    cvt_kernel<<<dim3((n8x + 255) / 256), 256, 0, stream>>>(x,  xb, n8x);
    int n8w = D * D / 8;
    cvt_kernel<<<dim3((n8w + 255) / 256), 256, 0, stream>>>(Wk, Wb,               n8w);
    cvt_kernel<<<dim3((n8w + 255) / 256), 256, 0, stream>>>(Wq, Wb + D * D,       n8w);
    cvt_kernel<<<dim3((n8w + 255) / 256), 256, 0, stream>>>(Wv, Wb + 2 * D * D,   n8w);

    qkv_proj_kernel<<<dim3((BATCH * SEQ) / 16, 4, 3), 128, 0, stream>>>(
        xb, Wb, bk, bq, bv, Kb, Qb, Vt);
    scores_kernel<<<dim3(SEQ / 16, 4, BATCH), 128, 0, stream>>>(Qb, Kb, attn);
    softmax_kernel<<<dim3(BATCH * SEQ), 256, 0, stream>>>(attn, Pb);
    av_kernel<<<dim3(SEQ / 16, 4, BATCH), 128, 0, stream>>>(Pb, Vt, out);
}
